// Attention_70205535420559
// MI455X (gfx1250) — matrix operand order repaired and checked by reading, not compiled, not run on hardware
//
#include <hip/hip_runtime.h>

// ---------------------------------------------------------------------------
// Types (all-trivial ext_vectors so unions are legal; bf16 storage kept as u16,
// only materialized as v16bf WMMA fragments)
// ---------------------------------------------------------------------------
typedef unsigned short u16;
typedef __attribute__((ext_vector_type(16))) __bf16 v16bf;
typedef __attribute__((ext_vector_type(8)))  float  v8f;
typedef __attribute__((ext_vector_type(4)))  float  f32x4;
typedef __attribute__((ext_vector_type(4)))  u16    u16x4;

union FragBF {
  v16bf v;
  f32x4 f4[2];
};

union BStage {          // 64B of staged B-tile data per thread
  f32x4 v[4];
  u16   s[32];
};

__device__ __forceinline__ u16 f32_to_bf16(float f) {
  unsigned int u = __float_as_uint(f);
  u += 0x7FFFu + ((u >> 16) & 1u);   // round-to-nearest-even
  return (u16)(u >> 16);
}

// ---------------------------------------------------------------------------
// f32 -> bf16 conversion, 4-wide (pure HBM bandwidth)
// ---------------------------------------------------------------------------
__global__ __launch_bounds__(256)
void cvt_f32_bf16_kernel(const f32x4* __restrict__ in, u16x4* __restrict__ out,
                         long n4) {
  long i = (long)blockIdx.x * blockDim.x + threadIdx.x;
  long stride = (long)gridDim.x * blockDim.x;
  for (; i < n4; i += stride) {
    f32x4 v = in[i];
    u16x4 o;
    o.x = f32_to_bf16(v.x);
    o.y = f32_to_bf16(v.y);
    o.z = f32_to_bf16(v.z);
    o.w = f32_to_bf16(v.w);
    out[i] = o;
  }
}

// ---------------------------------------------------------------------------
// Tiled bf16 WMMA GEMM: C[M,N] = A[M,K] @ B, batched (gridDim.z).
//   TRANS_B: B given as [N,K] row-major (compute A @ B^T)
//   EPI: 0 = store f32, 1 = store bf16, 2 = store f32 tanh(x + bias[col])
//   SPLIT_A: A columns [0,kSplit) from A0, [kSplit,K) from A1 (fused concat)
//
// Block: 256 threads = 8 waves in a 2(M) x 4(N) grid. Block tile 128x256,
// K-step 32. Each wave owns a 64x64 patch = 4x4 WMMA tiles -> 16
// v_wmma_f32_16x16x32_bf16 per K-step against 8 fragment ds_load pairs.
// Double-buffered LDS + register-staged global loads: next tile's global
// loads issue before the WMMA block, ds_stores drain into the spare buffer
// after it, one barrier per K-step.
// ---------------------------------------------------------------------------
#define BM 128
#define BN 256
#define BK 32
#define LDSK 40   // 32 + 8 pad: 80B row pitch, 16B aligned, bank-conflict-free

template<bool TRANS_B, int EPI, bool SPLIT_A>
__global__ __launch_bounds__(256)
void gemm_bf16_kernel(const u16* __restrict__ A0, const u16* __restrict__ A1,
                      int kSplit, const u16* __restrict__ Bm,
                      void* __restrict__ Cv, const float* __restrict__ bias,
                      int M, int N, int K, int lda, int ldb, int ldc,
                      long sA, long sB, long sC) {
  __shared__ alignas(16) u16 As[2][BM][LDSK];
  __shared__ alignas(16) u16 Bs[2][BN][LDSK];   // stored [n][k] (B transposed)

  const int tid   = threadIdx.x;
  const int lane  = tid & 31;
  const int wave  = tid >> 5;
  const int waveM = (wave >> 2) * 64;   // 0 / 64
  const int waveN = (wave & 3) * 64;    // 0 / 64 / 128 / 192
  const int blockN = blockIdx.x * BN;
  const int blockM = blockIdx.y * BM;
  const long batch = blockIdx.z;

  const u16* Abase0 = A0 + batch * sA;
  const u16* Abase1 = SPLIT_A ? (A1 + batch * sA) : A0;
  const u16* Bbase  = Bm + batch * sB;

  v8f acc[4][4];
#pragma unroll
  for (int mi = 0; mi < 4; ++mi)
#pragma unroll
    for (int ni = 0; ni < 4; ++ni)
#pragma unroll
      for (int r = 0; r < 8; ++r) acc[mi][ni][r] = 0.0f;

  // ---- staging assignments ----
  const int arow = tid >> 1;          // A: 2 threads per row, 32B each
  const int acol = (tid & 1) * 16;
  const int brow = tid;               // B trans: 1 thread per n-row, 64B
  const int bk   = tid >> 3;          // B NN: k-line
  const int bn0  = (tid & 7) * 32;    // B NN: 32 n per thread

  f32x4  ra[2];
  BStage rb;

  auto loadA = [&](int kk) {
    const u16* src;
    if (SPLIT_A && kk >= kSplit)
      src = Abase1 + (long)(blockM + arow) * lda + (kk - kSplit) + acol;
    else
      src = Abase0 + (long)(blockM + arow) * lda + kk + acol;
    ra[0] = *(const f32x4*)(src);
    ra[1] = *(const f32x4*)(src + 8);
  };
  auto loadB = [&](int kk) {
    if (TRANS_B) {
      const u16* src = Bbase + (long)(blockN + brow) * ldb + kk;
#pragma unroll
      for (int i = 0; i < 4; ++i) rb.v[i] = *(const f32x4*)(src + i * 8);
    } else {
      const u16* src = Bbase + (long)(kk + bk) * ldb + blockN + bn0;
#pragma unroll
      for (int i = 0; i < 4; ++i) rb.v[i] = *(const f32x4*)(src + i * 8);
    }
  };
  auto storeAB = [&](int buf) {
    *(f32x4*)&As[buf][arow][acol]     = ra[0];
    *(f32x4*)&As[buf][arow][acol + 8] = ra[1];
    if (TRANS_B) {
#pragma unroll
      for (int i = 0; i < 4; ++i) *(f32x4*)&Bs[buf][brow][i * 8] = rb.v[i];
    } else {
#pragma unroll
      for (int i = 0; i < 32; ++i) Bs[buf][bn0 + i][bk] = rb.s[i];
    }
  };

  // fragment addressing (ISA 7.12.2: 16-bit A 16x32 / B 32x16 layouts)
  const int mrow = lane & 15;
  const int ka = (lane < 16) ? 0 : 8;   // A: lo lanes K=0-7,16-23; hi +8
  const int kb = (lane < 16) ? 0 : 8;   // B: lo lanes K=0-7,16-23; hi +8

  // ---- prologue: stage tile 0 ----
  loadA(0);
  loadB(0);
  storeAB(0);
  __syncthreads();

  const int nk = K / BK;
  for (int ik = 0; ik < nk; ++ik) {
    const int buf = ik & 1;
    const bool more = (ik + 1 < nk);
    if (more) {                 // issue next tile's global loads early
      loadA((ik + 1) * BK);
      loadB((ik + 1) * BK);
    }

    FragBF af[4], bfr[4];
#pragma unroll
    for (int mi = 0; mi < 4; ++mi) {
      const u16* p = &As[buf][waveM + mi * 16 + mrow][0];
      af[mi].f4[0] = *(const f32x4*)(p + ka);
      af[mi].f4[1] = *(const f32x4*)(p + ka + 16);
    }
#pragma unroll
    for (int ni = 0; ni < 4; ++ni) {
      const u16* p = &Bs[buf][waveN + ni * 16 + mrow][0];
      bfr[ni].f4[0] = *(const f32x4*)(p + kb);
      bfr[ni].f4[1] = *(const f32x4*)(p + kb + 16);
    }
#pragma unroll
    for (int mi = 0; mi < 4; ++mi)
#pragma unroll
      for (int ni = 0; ni < 4; ++ni)
        acc[mi][ni] = __builtin_amdgcn_wmma_f32_16x16x32_bf16(
            false, af[mi].v, false, bfr[ni].v, (short)0, acc[mi][ni],
            false, false);

    if (more) storeAB(buf ^ 1);   // drain staged regs into spare buffer
    __syncthreads();
  }

  // ---- epilogue: C/D layout lanes 0-15 N=lane M=r, lanes 16-31 M=r+8 ----
  const int mrow2 = (lane < 16) ? 0 : 8;
  const int ncol  = lane & 15;
#pragma unroll
  for (int mi = 0; mi < 4; ++mi) {
#pragma unroll
    for (int ni = 0; ni < 4; ++ni) {
      const int col = blockN + waveN + ni * 16 + ncol;
#pragma unroll
      for (int r = 0; r < 8; ++r) {
        const int row = blockM + waveM + mi * 16 + mrow2 + r;
        const float v = acc[mi][ni][r];
        if (EPI == 0) {
          float* C = (float*)Cv + batch * sC;
          C[(long)row * ldc + col] = v;
        } else if (EPI == 1) {
          u16* C = (u16*)Cv + batch * sC;
          C[(long)row * ldc + col] = f32_to_bf16(v);
        } else {
          float* C = (float*)Cv + batch * sC;
          C[(long)row * ldc + col] = tanhf(v + bias[col]);
        }
      }
    }
  }
}

// ---------------------------------------------------------------------------
// Masked row softmax over Ts; one 256-thread block per (b,t) row.
// mask: s < count(source[b,:] != 0)   (TF sequence_mask semantics)
// ---------------------------------------------------------------------------
__global__ __launch_bounds__(256)
void softmax_mask_kernel(const float* __restrict__ score,
                         const int* __restrict__ source,
                         u16* __restrict__ aout, int Tt, int Ts) {
  const int t = blockIdx.x, b = blockIdx.y, tid = threadIdx.x;
  const float* row = score + ((long)b * Tt + t) * Ts;
  u16* arow = aout + ((long)b * Tt + t) * Ts;
  const int* src = source + (long)b * Ts;

  __shared__ float red[256];

  int cnt = 0;
  for (int s = tid; s < Ts; s += 256) cnt += (src[s] != 0);
  red[tid] = (float)cnt;
  __syncthreads();
  for (int off = 128; off > 0; off >>= 1) {
    if (tid < off) red[tid] += red[tid + off];
    __syncthreads();
  }
  const int len = (int)red[0];
  __syncthreads();

  float x[8];
  const int iters = Ts >> 8;    // Ts == 1024 -> 4
  float m = -3.4e38f;
  for (int i = 0; i < iters; ++i) {
    x[i] = row[tid + (i << 8)];
    m = fmaxf(m, x[i]);
  }
  red[tid] = m;
  __syncthreads();
  for (int off = 128; off > 0; off >>= 1) {
    if (tid < off) red[tid] = fmaxf(red[tid], red[tid + off]);
    __syncthreads();
  }
  m = red[0];
  __syncthreads();

  float sum = 0.f;
  for (int i = 0; i < iters; ++i) {
    const int s = tid + (i << 8);
    const float e = (s < len) ? __expf(x[i] - m) : 0.f;
    x[i] = e;
    sum += e;
  }
  red[tid] = sum;
  __syncthreads();
  for (int off = 128; off > 0; off >>= 1) {
    if (tid < off) red[tid] += red[tid + off];
    __syncthreads();
  }
  const float inv = 1.0f / red[0];
  for (int i = 0; i < iters; ++i)
    arow[tid + (i << 8)] = f32_to_bf16(x[i] * inv);
}

// ---------------------------------------------------------------------------
// Orchestration
// ---------------------------------------------------------------------------
extern "C" void kernel_launch(void* const* d_in, const int* in_sizes, int n_in,
                              void* d_out, int out_size, void* d_ws,
                              size_t ws_size, hipStream_t stream) {
  (void)in_sizes; (void)n_in; (void)out_size; (void)ws_size;

  constexpr int B = 16, T = 1024, H = 1024, O = 1024;
  constexpr long NTH = (long)B * T * H;   // 16M elements

  const float* ht   = (const float*)d_in[0];
  const float* hs   = (const float*)d_in[1];
  const float* Wa   = (const float*)d_in[2];
  const float* Wc   = (const float*)d_in[3];
  const float* bias = (const float*)d_in[4];
  const int* source = (const int*)d_in[5];
  float* out = (float*)d_out;

  char* ws = (char*)d_ws;
  size_t off = 0;
  u16* htb    = (u16*)(ws + off); off += (size_t)NTH * 2;           // 32 MB
  u16* hsb    = (u16*)(ws + off); off += (size_t)NTH * 2;           // 32 MB
  u16* Wab    = (u16*)(ws + off); off += (size_t)H * H * 2;         //  2 MB
  u16* Wcb    = (u16*)(ws + off); off += (size_t)2 * H * O * 2;     //  4 MB
  u16* hsproj = (u16*)(ws + off); off += (size_t)NTH * 2;           // 32 MB
  float* score = (float*)(ws + off); off += (size_t)B * T * T * 4;  // 64 MB
  u16* amat   = (u16*)(ws + off); off += (size_t)B * T * T * 2;     // 32 MB
  u16* cmat   = hsproj;   // hs_proj fully consumed before c is produced

  // 1) downcast inputs to bf16
  cvt_f32_bf16_kernel<<<2048, 256, 0, stream>>>(
      (const f32x4*)ht, (u16x4*)htb, NTH / 4);
  cvt_f32_bf16_kernel<<<2048, 256, 0, stream>>>(
      (const f32x4*)hs, (u16x4*)hsb, NTH / 4);
  cvt_f32_bf16_kernel<<<512, 256, 0, stream>>>(
      (const f32x4*)Wa, (u16x4*)Wab, (long)H * H / 4);
  cvt_f32_bf16_kernel<<<1024, 256, 0, stream>>>(
      (const f32x4*)Wc, (u16x4*)Wcb, (long)2 * H * O / 4);

  dim3 blk(256);
  dim3 grd(T / BN, T / BM, B);    // N/256, M/128, batch

  // 2) hs_proj[b] = hs[b] @ W_a            (bf16 out)
  gemm_bf16_kernel<false, 1, false><<<grd, blk, 0, stream>>>(
      hsb, nullptr, 0, Wab, hsproj, nullptr,
      T, H, H, H, H, H, (long)T * H, 0L, (long)T * H);

  // 3) score[b] = ht[b] @ hs_proj[b]^T     (f32 out, NT case)
  gemm_bf16_kernel<true, 0, false><<<grd, blk, 0, stream>>>(
      htb, nullptr, 0, hsproj, score, nullptr,
      T, T, H, H, H, T, (long)T * H, (long)T * H, (long)T * T);

  // 4) a = masked softmax(score)           (bf16 out)
  softmax_mask_kernel<<<dim3(T, B), blk, 0, stream>>>(score, source, amat, T, T);

  // 5) c[b] = a[b] @ hs[b]                 (bf16 out)
  gemm_bf16_kernel<false, 1, false><<<grd, blk, 0, stream>>>(
      amat, nullptr, 0, hsb, cmat, nullptr,
      T, H, T, T, H, H, (long)T * T, (long)T * H, (long)T * H);

  // 6) out[b] = tanh([c[b] | ht[b]] @ W_c + bias)   (fused concat + epilogue)
  gemm_bf16_kernel<false, 2, true><<<grd, blk, 0, stream>>>(
      cmat, htb, H, Wcb, out, bias,
      T, O, 2 * H, H, O, O, (long)T * H, 0L, (long)T * O);
}
